// TAdaConv2d_48249662604038
// MI455X (gfx1250) — matrix operand order repaired and checked by reading, not compiled, not run on hardware
//
#include <hip/hip_runtime.h>

typedef __attribute__((ext_vector_type(16))) __bf16 v16bf;
typedef __attribute__((ext_vector_type(8)))  __bf16 v8bf;
typedef __attribute__((ext_vector_type(8)))  float  v8f;

#define C_IN   256
#define C_OUT  256
#define T_DIM  8
#define H_DIM  56
#define W_DIM  56
#define PIX    (H_DIM * W_DIM)          // 3136
#define NIMG   32                        // B*T
#define CH_STRIDE (T_DIM * PIX)          // 25088 (x: stride between channels)

// workspace layout (in __bf16 elements)
#define GUARD_ELEMS 512                               // zeroed pad-read region (1 KB)
#define XS_ELEMS    (NIMG * PIX * C_IN)               // 25,690,112  (NHWC bf16 activations)
#define WPK_OFF     (GUARD_ELEMS + XS_ELEMS)          // packed weights after xs
#define WPK_ELEMS   (9 * 8 * C_OUT * 32)              // 589,824

// ---------------------------------------------------------------------------
// zero the pad-guard region
__global__ void tada_guard_zero(unsigned int* ws_u32) {
    ws_u32[threadIdx.x] = 0u;   // 256 threads * 4B = 1024B = GUARD_ELEMS bf16
}

// ---------------------------------------------------------------------------
// Pass 1: xs[img][pix][c] = bf16( x[b][c][t][pix] * alpha[b][c][t] )   (NHWC)
__global__ void tada_scale_nhwc(const float* __restrict__ x,
                                const float* __restrict__ alpha,
                                __bf16* __restrict__ ws) {
    const int img = blockIdx.y;               // 0..31
    const int b = img >> 3, t = img & 7;
    const int p = blockIdx.x * 256 + threadIdx.x;
    if (p >= PIX) return;

    const float* xp = x + (size_t)(b * C_IN) * CH_STRIDE + (size_t)t * PIX + p;
    const float* ap = alpha + (b * C_IN) * T_DIM + t;
    __bf16* op = ws + GUARD_ELEMS + ((size_t)img * PIX + p) * C_IN;

    for (int c0 = 0; c0 < C_IN; c0 += 8) {
        v8bf v;
#pragma unroll
        for (int i = 0; i < 8; ++i) {
            int c = c0 + i;
            v[i] = (__bf16)(xp[(size_t)c * CH_STRIDE] * ap[c * T_DIM]);
        }
        *(v8bf*)(op + c0) = v;   // 16B store; lines merge in L2
    }
}

// ---------------------------------------------------------------------------
// Pass 2: pack weights bf16 in WMMA A-fragment order.
// wpk[((kk*8+cb)*256 + m)*32 + j]; slot j -> kc = j (contiguous) so that
// lane<16 reads elems [0..15] = K{0..15}, lane>=16 reads [16..31] = K{16..31}
__global__ void tada_wpack(const float* __restrict__ w, __bf16* __restrict__ wpk) {
    const int o = blockIdx.x * 256 + threadIdx.x;
    if (o >= WPK_ELEMS) return;
    const int j  = o & 31;
    const int m  = (o >> 5) & 255;
    const int cb = (o >> 13) & 7;
    const int kk = o >> 16;                       // 0..8
    const int s = j >> 3, r = j & 7;
    const int kc = (s << 3) + r;   // kc = j (contiguous)
    const int ci = cb * 32 + kc;
    wpk[o] = (__bf16)(w[(m * C_IN + ci) * 9 + kk]);
}

// ---------------------------------------------------------------------------
// Conv: implicit GEMM, D[64 x 64pix] per block; 8 waves = 4(M) x 2(row-pairs)
__global__ void __launch_bounds__(256)
tada_conv_wmma(const __bf16* __restrict__ ws,
               const float* __restrict__ bias,
               float* __restrict__ out) {
    const int img = blockIdx.z;                 // 0..31
    const int b = img >> 3, t = img & 7;
    const int m0 = blockIdx.y * 64;             // c_out tile
    const int ct = blockIdx.x & 3;              // col tile (16 cols, last partial)
    const int rt = blockIdx.x >> 2;             // row tile (4 rows), 0..13
    const int h0 = rt * 4, w0 = ct * 16;

    const int wave = threadIdx.x >> 5;
    const int lane = threadIdx.x & 31;
    const int msub = wave & 3;                  // which 16-row M subtile
    const int nsel = wave >> 2;                 // which pair of output rows
    const int ln = lane & 15;
    const int lh = lane >> 4;
    const int lh16 = lh * 16;

    const int mrow = m0 + msub * 16 + ln;       // A-fragment row for this lane
    const int xsbase = GUARD_ELEMS + img * (PIX * C_IN);
    const int hb = h0 + nsel * 2;

    v8f acc0 = {}, acc1 = {};

    for (int kk = 0; kk < 9; ++kk) {
        const int dh = kk / 3, dw = kk - dh * 3;
        const int iw  = w0 + ln + dw - 1;
        const int ih0 = hb + dh - 1;
        const bool vw = (unsigned)iw < (unsigned)W_DIM;
        // branchless pad: invalid lanes read from zeroed guard (EXEC stays all-1s)
        const int off0 = (vw && (unsigned)ih0 < (unsigned)H_DIM)
                             ? xsbase + (ih0 * W_DIM + iw) * C_IN + lh16 : lh16;
        const int off1 = (vw && (unsigned)(ih0 + 1) < (unsigned)H_DIM)
                             ? xsbase + ((ih0 + 1) * W_DIM + iw) * C_IN + lh16 : lh16;
        const __bf16* aP = ws + WPK_OFF + ((kk * 8) * 256 + mrow) * 32 + lh16;

#pragma unroll
        for (int cb = 0; cb < 8; ++cb) {
            v16bf a  = *(const v16bf*)(aP + cb * 8192);          // 256*32 per cb
            v16bf b0 = *(const v16bf*)(ws + off0 + cb * 32);
            v16bf b1 = *(const v16bf*)(ws + off1 + cb * 32);
            acc0 = __builtin_amdgcn_wmma_f32_16x16x32_bf16(
                       false, a, false, b0, (short)0, acc0, false, false);
            acc1 = __builtin_amdgcn_wmma_f32_16x16x32_bf16(
                       false, a, false, b1, (short)0, acc1, false, false);
        }
    }

    // store C/D: VGPR r -> M = r + 8*lh, N = ln  (ISA 7.12.2 f32 16x16 layout)
    const int wcol = w0 + ln;
    if (wcol < W_DIM) {
#pragma unroll
        for (int r = 0; r < 8; ++r) {
            const int m = m0 + msub * 16 + lh * 8 + r;
            const float bb = bias[m];
            const size_t obase = ((size_t)(b * C_OUT + m) * T_DIM + t) * PIX + wcol;
            out[obase + (size_t)(hb + 0) * W_DIM] = acc0[r] + bb;
            out[obase + (size_t)(hb + 1) * W_DIM] = acc1[r] + bb;
        }
    }
}

// ---------------------------------------------------------------------------
extern "C" void kernel_launch(void* const* d_in, const int* in_sizes, int n_in,
                              void* d_out, int out_size, void* d_ws, size_t ws_size,
                              hipStream_t stream) {
    const float* x      = (const float*)d_in[0];
    const float* alpha  = (const float*)d_in[1];
    const float* weight = (const float*)d_in[2];
    const float* bias   = (const float*)d_in[3];
    float* out = (float*)d_out;
    __bf16* ws = (__bf16*)d_ws;

    tada_guard_zero<<<1, 256, 0, stream>>>((unsigned int*)d_ws);

    dim3 g1((PIX + 255) / 256, NIMG);
    tada_scale_nhwc<<<g1, 256, 0, stream>>>(x, alpha, ws);

    tada_wpack<<<(WPK_ELEMS + 255) / 256, 256, 0, stream>>>(weight, ws + WPK_OFF);

    dim3 gc(56 /*4 col x 14 row tiles*/, C_OUT / 64, NIMG);
    tada_conv_wmma<<<gc, 256, 0, stream>>>(ws, bias, out);
}
